// SPR_RGCN_88648124991102
// MI455X (gfx1250) — matrix operand order repaired and checked by reading, not compiled, not run on hardware
//
#include <hip/hip_runtime.h>

#define HID 64
#define NND 100000
#define NED 1600000
#define NREL 3
#define NGRAPH 1024
#define NCLS 10

typedef _Float16 v16h __attribute__((ext_vector_type(16)));
typedef _Float16 h8   __attribute__((ext_vector_type(8)));
typedef _Float16 h4   __attribute__((ext_vector_type(4)));
typedef float    v8f  __attribute__((ext_vector_type(8)));

// ---------------- utility ----------------
__global__ void k_zero(float* __restrict__ p, long n) {
    long i = (long)blockIdx.x * blockDim.x + threadIdx.x;
    if (i < n) p[i] = 0.0f;
}

// per-(relation,dst) in-degree counts
__global__ void k_count(const int* __restrict__ ei, const int* __restrict__ et,
                        float* __restrict__ cnt) {
    int e = blockIdx.x * blockDim.x + threadIdx.x;
    if (e >= NED) return;
    int d = ei[NED + e];         // edge_index[1][e]
    int r = et[e];
    atomicAdd(&cnt[(long)r * NND + d], 1.0f);
}

__global__ void k_invert(float* __restrict__ cnt, long n) {
    long i = (long)blockIdx.x * blockDim.x + threadIdx.x;
    if (i < n) cnt[i] = 1.0f / fmaxf(cnt[i], 1.0f);
}

// ---------------- weight pre-pack into WMMA B-fragment layout ----------------
// bp[t], t = (((m*4 + nt)*2 + kt)*32 + lane)*16 + i
//   = (f16) W_m[(kt*32 + 16*(lane>>4) + i)*HID + nt*16 + (lane&15)]
// One fragment (m,nt,kt) = 32 lanes x 16 halves = 1 KB; total 4*4*2 = 32 KB.
__global__ void k_pack_w(const float* __restrict__ wrel, const float* __restrict__ wroot,
                         _Float16* __restrict__ bp) {
    int t = blockIdx.x * blockDim.x + threadIdx.x;
    if (t >= 4 * 4 * 2 * 32 * 16) return;
    int i    = t & 15;
    int lane = (t >> 4) & 31;
    int kt   = (t >> 9) & 1;
    int nt   = (t >> 10) & 3;
    int m    = t >> 12;                 // 0..2 relations, 3 = root
    int half = lane >> 4, l = lane & 15;
    int K    = kt * 32 + 16 * half + i;
    int col  = nt * 16 + l;
    const float* wp = (m < 3) ? (wrel + (size_t)m * HID * HID) : wroot;
    bp[t] = (_Float16)wp[K * HID + col];
}

// ---------------- embed + lin0 + relu -> f16 features ----------------
__global__ void k_embed_lin0(const int* __restrict__ xidx,
                             const float* __restrict__ semb, const float* __restrict__ cemb,
                             const float* __restrict__ w, const float* __restrict__ b,
                             _Float16* __restrict__ xA) {
    long t = (long)blockIdx.x * blockDim.x + threadIdx.x;
    if (t >= (long)NND * HID) return;
    int n = (int)(t >> 6), c = (int)(t & 63);
    int i0 = xidx[2 * n], i1 = xidx[2 * n + 1];
    const float* s  = semb + i0 * 8;
    const float* cc = cemb + i1 * 8;
    float acc = b[c];
#pragma unroll
    for (int k = 0; k < 8; k++) acc += s[k]  * w[k * HID + c];
#pragma unroll
    for (int k = 0; k < 8; k++) acc += cc[k] * w[(8 + k) * HID + c];
    xA[t] = (_Float16)fmaxf(acc, 0.0f);
}

// ---------------- WMMA GEMM: xw[r] = X*W_r (f16 out), acc = X*W_root + b ----
// One block = one 16-node tile. Waves 0..2 -> relations, wave 3 -> root.
// Layouts (wave32); A/B k order is this code's own, C/D as in ISA 7.12.2:
//   A f16 16x32 (contiguous k, chosen here): lanes L<16 row M=L hold K[0..15]; lanes>=16 hold K[16..31]
//   B f16 32x16 (contiguous k, matching A): lanes L<16 col N=L hold K[0..15]; lanes>=16 hold K[16..31]
//   C/D f32 16x16: vgpr j -> M = j + 8*(lane>=16), N = lane%16
// B fragments come pre-packed (k_pack_w): one aligned 32B vector load each.
__global__ void __launch_bounds__(128)
k_rgcn_gemm(const _Float16* __restrict__ xA,
            const _Float16* __restrict__ bp,     // packed fragments for this layer
            const float* __restrict__ bias,
            _Float16* __restrict__ xw, float* __restrict__ acc) {
    const int mt   = blockIdx.x;          // node tile
    const int wave = threadIdx.x >> 5;    // matrix id 0..3
    const int lane = threadIdx.x & 31;
    const int half = lane >> 4;
    const int l    = lane & 15;
    const int row  = mt * 16 + l;

    // A: two K-halves of 32, two 16B loads each
    v16h a0, a1;
    {
        const _Float16* xr = xA + (size_t)row * HID;
        h8 lo0 = *(const h8*)(xr + 0  + 16 * half);
        h8 hi0 = *(const h8*)(xr + 8  + 16 * half);
        h8 lo1 = *(const h8*)(xr + 32 + 16 * half);
        h8 hi1 = *(const h8*)(xr + 40 + 16 * half);
#pragma unroll
        for (int i = 0; i < 8; i++) {
            a0[i] = lo0[i]; a0[i + 8] = hi0[i];
            a1[i] = lo1[i]; a1[i + 8] = hi1[i];
        }
    }

    const _Float16* fb = bp + ((size_t)wave * 8 + 0) * 32 * 16 + lane * 16;

#pragma unroll
    for (int nt = 0; nt < 4; ++nt) {
        const int no = nt * 16;
        v8f c = {};
        v16h b0 = *(const v16h*)(fb + (nt * 2 + 0) * 32 * 16);   // K = 0..31
        v16h b1 = *(const v16h*)(fb + (nt * 2 + 1) * 32 * 16);   // K = 32..63
        c = __builtin_amdgcn_wmma_f32_16x16x32_f16(false, a0, false, b0, (short)0, c, false, false);
        c = __builtin_amdgcn_wmma_f32_16x16x32_f16(false, a1, false, b1, (short)0, c, false, false);

        if (wave < 3) {
#pragma unroll
            for (int j = 0; j < 8; j++) {
                int m = mt * 16 + j + 8 * half;
                xw[((size_t)wave * NND + m) * HID + no + l] = (_Float16)c[j];
            }
        } else {
            float bv = bias[no + l];
#pragma unroll
            for (int j = 0; j < 8; j++) {
                int m = mt * 16 + j + 8 * half;
                acc[(size_t)m * HID + no + l] = c[j] + bv;
            }
        }
    }
}

// ---------------- edge scatter: acc[d] += xw[r][s] * inv_cnt[r][d] ----------
// 16 threads per edge, 4 channels each (8B f16 gather, 4 f32 atomics).
__global__ void k_edge_scatter(const int* __restrict__ ei, const int* __restrict__ et,
                               const _Float16* __restrict__ xw, const float* __restrict__ invc,
                               float* __restrict__ acc) {
    long t = (long)blockIdx.x * blockDim.x + threadIdx.x;
    if (t >= (long)NED * 16) return;
    int e = (int)(t >> 4);
    int c = ((int)t & 15) * 4;
    int s = ei[e], d = ei[NED + e], r = et[e];
    float sc = invc[(long)r * NND + d];
    h4 v = *(const h4*)(xw + ((size_t)r * NND + s) * HID + c);
    float* dst = acc + (size_t)d * HID + c;
    atomicAdd(dst + 0, (float)v[0] * sc);
    atomicAdd(dst + 1, (float)v[1] * sc);
    atomicAdd(dst + 2, (float)v[2] * sc);
    atomicAdd(dst + 3, (float)v[3] * sc);
}

// ---------------- relu in place + f16 copy for next GEMM ----------------
__global__ void k_relu_cvt(float* __restrict__ acc, _Float16* __restrict__ xA) {
    long t = (long)blockIdx.x * blockDim.x + threadIdx.x;
    if (t >= (long)NND * HID) return;
    float v = fmaxf(acc[t], 0.0f);
    acc[t] = v;
    xA[t] = (_Float16)v;
}

// ---------------- global mean pool (sums + counts) ----------------
__global__ void k_pool(const float* __restrict__ acc, const int* __restrict__ batch,
                       float* __restrict__ pooled, float* __restrict__ gcnt) {
    long t = (long)blockIdx.x * blockDim.x + threadIdx.x;
    if (t >= (long)NND * HID) return;
    int n = (int)(t >> 6), h = (int)(t & 63);
    int g = batch[n];
    atomicAdd(&pooled[(long)g * HID + h], acc[t]);
    if (h == 0) atomicAdd(&gcnt[g], 1.0f);
}

// ---------------- classifier ----------------
__global__ void k_cls(const float* __restrict__ pooled, const float* __restrict__ gcnt,
                      const float* __restrict__ w, const float* __restrict__ b,
                      float* __restrict__ out) {
    int t = blockIdx.x * blockDim.x + threadIdx.x;
    if (t >= NGRAPH * NCLS) return;
    int g = t / NCLS, c = t % NCLS;
    float inv = 1.0f / fmaxf(gcnt[g], 1.0f);
    float acc = b[c];
#pragma unroll
    for (int h = 0; h < HID; h++) acc += pooled[(long)g * HID + h] * inv * w[h * NCLS + c];
    out[t] = acc;
}

extern "C" void kernel_launch(void* const* d_in, const int* in_sizes, int n_in,
                              void* d_out, int out_size, void* d_ws, size_t ws_size,
                              hipStream_t stream) {
    (void)in_sizes; (void)n_in; (void)out_size; (void)ws_size;
    const int*   x_idx   = (const int*)d_in[0];
    const int*   ei      = (const int*)d_in[1];
    const int*   et      = (const int*)d_in[2];
    const int*   batch   = (const int*)d_in[3];
    const float* semb    = (const float*)d_in[4];
    const float* cemb    = (const float*)d_in[5];
    const float* lin0_w  = (const float*)d_in[6];
    const float* lin0_b  = (const float*)d_in[7];
    const float* r1_wrel = (const float*)d_in[8];
    const float* r1_wroot= (const float*)d_in[9];
    const float* r1_b    = (const float*)d_in[10];
    const float* r2_wrel = (const float*)d_in[11];
    const float* r2_wroot= (const float*)d_in[12];
    const float* r2_b    = (const float*)d_in[13];
    const float* cls_w   = (const float*)d_in[14];
    const float* cls_b   = (const float*)d_in[15];
    float* out = (float*)d_out;

    // workspace carve-out (all L2-resident on MI455X: ~78 MB total)
    char* ws = (char*)d_ws;
    size_t off = 0;
    auto carve = [&](size_t bytes) -> char* {
        char* p = ws + off;
        off = (off + bytes + 255) & ~(size_t)255;
        return p;
    };
    float*     invc   = (float*)    carve((size_t)NREL * NND * sizeof(float));          // 1.2 MB
    _Float16*  xA     = (_Float16*) carve((size_t)NND * HID * sizeof(_Float16));        // 12.8 MB
    _Float16*  xw     = (_Float16*) carve((size_t)NREL * NND * HID * sizeof(_Float16)); // 38.4 MB
    float*     acc    = (float*)    carve((size_t)NND * HID * sizeof(float));           // 25.6 MB
    float*     pooled = (float*)    carve((size_t)NGRAPH * HID * sizeof(float));
    float*     gcnt   = (float*)    carve((size_t)NGRAPH * sizeof(float));
    const int  NPACK  = 4 * 4 * 2 * 32 * 16;                                            // 16384
    _Float16*  bp1    = (_Float16*) carve((size_t)NPACK * sizeof(_Float16));            // 32 KB
    _Float16*  bp2    = (_Float16*) carve((size_t)NPACK * sizeof(_Float16));            // 32 KB

    const int T = 256;
    const long nNH = (long)NND * HID;

    // zero atomic accumulators (deterministic per call)
    k_zero<<<(NREL * NND + T - 1) / T, T, 0, stream>>>(invc, (long)NREL * NND);
    k_zero<<<(NGRAPH * HID + T - 1) / T, T, 0, stream>>>(pooled, (long)NGRAPH * HID);
    k_zero<<<(NGRAPH + T - 1) / T, T, 0, stream>>>(gcnt, (long)NGRAPH);

    // pre-pack both layers' weights into WMMA B-fragment layout (f16)
    k_pack_w<<<NPACK / T, T, 0, stream>>>(r1_wrel, r1_wroot, bp1);
    k_pack_w<<<NPACK / T, T, 0, stream>>>(r2_wrel, r2_wroot, bp2);

    // edge counts -> 1/max(cnt,1)   (shared by both layers)
    k_count<<<(NED + T - 1) / T, T, 0, stream>>>(ei, et, invc);
    k_invert<<<(NREL * NND + T - 1) / T, T, 0, stream>>>(invc, (long)NREL * NND);

    // embed + lin0 + relu
    k_embed_lin0<<<(int)((nNH + T - 1) / T), T, 0, stream>>>(x_idx, semb, cemb, lin0_w, lin0_b, xA);

    // ----- RGCN layer 1 -----
    k_rgcn_gemm<<<NND / 16, 128, 0, stream>>>(xA, bp1, r1_b, xw, acc);
    k_edge_scatter<<<(int)(((long)NED * 16 + T - 1) / T), T, 0, stream>>>(ei, et, xw, invc, acc);
    k_relu_cvt<<<(int)((nNH + T - 1) / T), T, 0, stream>>>(acc, xA);

    // ----- RGCN layer 2 -----
    k_rgcn_gemm<<<NND / 16, 128, 0, stream>>>(xA, bp2, r2_b, xw, acc);
    k_edge_scatter<<<(int)(((long)NED * 16 + T - 1) / T), T, 0, stream>>>(ei, et, xw, invc, acc);
    k_relu_cvt<<<(int)((nNH + T - 1) / T), T, 0, stream>>>(acc, xA);

    // mean-pool + classify
    k_pool<<<(int)((nNH + T - 1) / T), T, 0, stream>>>(acc, batch, pooled, gcnt);
    k_cls<<<(NGRAPH * NCLS + T - 1) / T, T, 0, stream>>>(pooled, gcnt, cls_w, cls_b, out);
}
